// ARCStacked_10557029614330
// MI455X (gfx1250) — matrix operand order repaired and checked by reading, not compiled, not run on hardware
//
#include <hip/hip_runtime.h>
#include <stdint.h>

#define BATCH 2
#define NHEAD 16
#define TSEQ  2304   // 2048 + 2*128
#define DIN   2048
#define DKV   128
#define WST   128

typedef __attribute__((ext_vector_type(16))) __bf16 v16bf;
typedef __attribute__((ext_vector_type(8)))  float  v8f;

union Frag16 { v16bf v; unsigned int u[8]; unsigned short h[16]; };
union FragC  { v8f   v; float f[8]; };
union U4     { uint4 q; unsigned int u[4]; unsigned short s[8]; };

struct CvtArgs  { const float* src[6]; unsigned short* dst[6]; };
struct ProjArgs { const unsigned short* wc[3]; const unsigned short* ws[3]; unsigned short* ob[3]; };

__device__ __forceinline__ unsigned short f2bf(float x) {
  unsigned u = __float_as_uint(x);
  return (unsigned short)((u + 0x7FFFu + ((u >> 16) & 1u)) >> 16);  // RNE
}

__device__ __forceinline__ v8f wmma_bf16(v16bf a, v16bf b, v8f c) {
  return __builtin_amdgcn_wmma_f32_16x16x32_bf16(false, a, false, b, (short)0, c, false, false);
}

// CDNA5 async global->LDS copy (ASYNCcnt), GVS form: uniform SGPR base +
// 32-bit per-lane byte offset (no per-lane 64-bit address math).
__device__ __forceinline__ void async_ld_b128_s(void* lds_dst, const void* sbase, unsigned voff) {
  unsigned ldsoff = (unsigned)(unsigned long long)lds_dst;   // low 32b of generic = LDS byte addr
  asm volatile("global_load_async_to_lds_b128 %0, %1, %2"
               :: "v"(ldsoff), "v"(voff), "s"(sbase) : "memory");
}
__device__ __forceinline__ void wait_async() {
  asm volatile("s_wait_asynccnt 0x0" ::: "memory");
}

// ---------------------------------------------------------------- stage 1a: x fp32->bf16 (vectorized)
__global__ __launch_bounds__(256)
void cvt_bf16_kernel(const float* __restrict__ src, unsigned short* __restrict__ dst, size_t n8) {
  size_t i = (size_t)blockIdx.x * blockDim.x + threadIdx.x;
  size_t stride = (size_t)gridDim.x * blockDim.x;
  for (; i < n8; i += stride) {
    const float4* s = (const float4*)(src + i * 8);
    float4 a = s[0], b = s[1];
    U4 o;
    o.s[0] = f2bf(a.x); o.s[1] = f2bf(a.y); o.s[2] = f2bf(a.z); o.s[3] = f2bf(a.w);
    o.s[4] = f2bf(b.x); o.s[5] = f2bf(b.y); o.s[6] = f2bf(b.z); o.s[7] = f2bf(b.w);
    ((uint4*)dst)[i] = o.q;
  }
}

// ---------------------------------------------------------------- stage 1b: all 6 weights -> bf16, transposed
// src f32 [H][D][E] -> dst bf16 [H][E][D]; 32x32 LDS-tiled so both sides coalesce.
__global__ __launch_bounds__(256)
void cvt_wT_kernel(CvtArgs ca) {
  __shared__ float tile[32][33];
  const float* __restrict__ src = ca.src[blockIdx.z];
  unsigned short* __restrict__ dst = ca.dst[blockIdx.z];
  const int h = blockIdx.y;
  const int td = (blockIdx.x & (DIN / 32 - 1)) * 32;   // d tile
  const int te = (blockIdx.x / (DIN / 32)) * 32;       // e tile
  const int r = threadIdx.x >> 5, c = threadIdx.x & 31;
#pragma unroll
  for (int i = 0; i < 4; ++i) {
    int row = r + i * 8;                                // d index
    tile[row][c] = src[((size_t)h * DIN + td + row) * DKV + te + c];
  }
  __syncthreads();
#pragma unroll
  for (int i = 0; i < 4; ++i) {
    int row = r + i * 8;                                // e index
    dst[((size_t)h * DKV + te + row) * DIN + td + c] = f2bf(tile[c][row]);
  }
}

// ---------------------------------------------------------------- stage 2
// All three projections in ONE launch (grid.y = 3*NHEAD): out = l2norm(x @ wT).
// Wave tiling 32x64, double-buffered async staging with XOR-toggled buffer
// offset (single loop body -> accumulators coalesce, no duplicated bodies).
__global__ __launch_bounds__(256)
void proj_l2_kernel(const unsigned short* __restrict__ xb,    // [B][T][D] bf16
                    ProjArgs pa)
{
  __shared__ unsigned short lxf[2 * 4096];  // x tiles  [buf][row t][k]  2x8KB
  __shared__ unsigned short lwf[2 * 4096];  // w tiles  [buf][row e][k]  2x8KB

  const int qt = blockIdx.x;
  const int pidx = blockIdx.y / NHEAD;      // 0=q 1=k 2=v
  const int h = blockIdx.y % NHEAD;
  const int b = blockIdx.z;
  const int tid = threadIdx.x;
  const int wv = tid >> 5, lane = tid & 31, lo = lane & 15, hi = lane >> 4;
  const int wr = wv >> 1;                   // row-tile pair 0..3  (rows 32*wr..+31)
  const int wc = wv & 1;                    // col half 0..1       (cols 64*wc..+63)
  const int t0 = qt * 128;
  const bool state = (qt == 0) || (qt == TSEQ / 128 - 1);
  const unsigned short* wrow = (state ? pa.ws[pidx] : pa.wc[pidx]) + (size_t)h * DKV * DIN;
  const unsigned short* xrow = xb + ((size_t)b * TSEQ + t0) * DIN;

  // loop-invariant per-lane staging offsets
  const int srow0 = tid >> 2,         sseg0 = (tid & 3) * 8;
  const int srow1 = (tid + 256) >> 2, sseg1 = ((tid + 256) & 3) * 8;
  const unsigned voff0 = (unsigned)(srow0 * DIN + sseg0) * 2u;
  const unsigned voff1 = (unsigned)(srow1 * DIN + sseg1) * 2u;

  FragC acc[2][4];
#pragma unroll
  for (int r = 0; r < 2; ++r)
#pragma unroll
    for (int c = 0; c < 4; ++c)
#pragma unroll
      for (int j = 0; j < 8; ++j) acc[r][c].f[j] = 0.0f;

  auto stage = [&](unsigned bufoff, int kk) {
    const unsigned short* xs = xrow + kk;
    const unsigned short* ws = wrow + kk;
    async_ld_b128_s(&lxf[bufoff + srow0 * 32 + sseg0], xs, voff0);
    async_ld_b128_s(&lwf[bufoff + srow0 * 32 + sseg0], ws, voff0);
    async_ld_b128_s(&lxf[bufoff + srow1 * 32 + sseg1], xs, voff1);
    async_ld_b128_s(&lwf[bufoff + srow1 * 32 + sseg1], ws, voff1);
  };

  stage(0, 0);
  wait_async();
  __syncthreads();

  unsigned cb = 0;                          // current buffer offset (elements)
  for (int kt = 0; kt < DIN / 32; ++kt) {
    const bool morek = (kt + 1 < DIN / 32);
    if (morek) stage(cb ^ 4096u, (kt + 1) * 32);

    Frag16 a0, a1;
#pragma unroll
    for (int j = 0; j < 8; ++j) {
      int k = ((j & 4) << 2) + hi * 8 + ((j & 3) << 1);
      a0.u[j] = *(const unsigned int*)&lxf[cb + (wr * 32 + lo) * 32 + k];
      a1.u[j] = *(const unsigned int*)&lxf[cb + (wr * 32 + 16 + lo) * 32 + k];
    }
#pragma unroll
    for (int c = 0; c < 4; ++c) {
      Frag16 bw;
      int n = wc * 64 + c * 16 + lo;
#pragma unroll
      for (int j = 0; j < 8; ++j)
        bw.u[j] = *(const unsigned int*)&lwf[cb + n * 32 + ((j & 4) << 2) + hi * 8 + ((j & 3) << 1)];
      acc[0][c].v = wmma_bf16(a0.v, bw.v, acc[0][c].v);
      acc[1][c].v = wmma_bf16(a1.v, bw.v, acc[1][c].v);
    }

    if (morek) {
      wait_async();
      __syncthreads();
    }
    cb ^= 4096u;
  }

  // L2 norm along 128 output cols: each wave holds 64 cols of its 32 rows; the
  // paired wave (wc^1) holds the other 64 -> combine partials through LDS.
  __shared__ float ssx[2][128];             // [wc][row] partial sums of squares
  float ss[2][8];
#pragma unroll
  for (int r = 0; r < 2; ++r)
#pragma unroll
    for (int j = 0; j < 8; ++j) {
      float s = 0.0f;
#pragma unroll
      for (int c = 0; c < 4; ++c) s += acc[r][c].f[j] * acc[r][c].f[j];
      ss[r][j] = s;
    }
#pragma unroll
  for (int r = 0; r < 2; ++r)
#pragma unroll
    for (int j = 0; j < 8; ++j)
      for (int mk = 1; mk < 16; mk <<= 1)
        ss[r][j] += __shfl_xor(ss[r][j], mk, 16);
#pragma unroll
  for (int r = 0; r < 2; ++r)
#pragma unroll
    for (int j = 0; j < 8; ++j)
      if (lo == 0) ssx[wc][wr * 32 + r * 16 + j + hi * 8] = ss[r][j];
  __syncthreads();

  unsigned short* ob = pa.ob[pidx] + (((size_t)b * NHEAD + h) * TSEQ + t0) * DKV;
#pragma unroll
  for (int r = 0; r < 2; ++r)
#pragma unroll
    for (int j = 0; j < 8; ++j) {
      int row = wr * 32 + r * 16 + j + hi * 8;
      float tot = ssx[0][row] + ssx[1][row];
      float inv = 1.0f / fmaxf(sqrtf(tot), 1e-12f);
#pragma unroll
      for (int c = 0; c < 4; ++c)
        ob[(size_t)row * DKV + wc * 64 + c * 16 + lo] = f2bf(acc[r][c].f[j] * inv);
    }
}

// ---------------------------------------------------------------- stage 3
// Flash attention. l2norm(q),l2norm(k) => |score*scale| <= 1/sqrt(128): softmax
// needs NO running max (p = exp(s*scale) directly). Row sums via ones-WMMA.
// qt reversed so the longest tiles launch first (work ~ 2*qt+2).
__global__ __launch_bounds__(256)
void attn_kernel(const unsigned short* __restrict__ qb,
                 const unsigned short* __restrict__ kbuf,
                 const unsigned short* __restrict__ vbuf,
                 const float* __restrict__ scaling,
                 float* __restrict__ out)
{
  __shared__ unsigned short lds[32768];                 // 64KB
  // K buffers: lds + buf*8192            ([64 key][128 d]);  P aliases dead K
  // V buffers: lds + 16384 + buf*8192    ([128 dv][64 key], transposed)

  const int qt = (TSEQ / 128 - 1) - blockIdx.x;         // longest-first
  const int h = blockIdx.y, b = blockIdx.z;
  const int tid = threadIdx.x;
  const int wv = tid >> 5, lane = tid & 31, lo = lane & 15, hi = lane >> 4;
  const float scl = scaling[h];
  const size_t headoff = ((size_t)b * NHEAD + h) * TSEQ;
  const unsigned short* qg = qb + (headoff + (size_t)qt * 128) * DKV;
  const unsigned short* kg = kbuf + headoff * DKV;
  const unsigned short* vg = vbuf + headoff * DKV;

  Frag16 ones;                                          // bf16 1.0 everywhere
#pragma unroll
  for (int j = 0; j < 8; ++j) ones.u[j] = 0x3F803F80u;

  // ---- stage Q tile [128][128] async into lds[0..16384)
#pragma unroll
  for (int i = 0; i < 8; ++i) {
    unsigned idx = (unsigned)(tid + 256 * i);           // 2048 b128 chunks
    async_ld_b128_s(&lds[idx * 8], qg, idx * 16u);
  }
  wait_async();
  __syncthreads();

  const int m = wv * 16 + lo;
  Frag16 qf[4];
#pragma unroll
  for (int kd = 0; kd < 4; ++kd)
#pragma unroll
    for (int j = 0; j < 8; ++j) {
      int k = kd * 32 + ((j & 4) << 2) + hi * 8 + ((j & 3) << 1);
      qf[kd].u[j] = *(const unsigned int*)&lds[m * 128 + k];
    }
  __syncthreads();   // done reading Q region before K overwrites it

  const int kb_lo = (qt == TSEQ / 128 - 1) ? 2 : 0;     // last q tile never sees keys [0,128)
  const int kb_hi = 2 * qt + 1;                         // causal block bound

  // ---- prologue: stage first K (async) + V (regs->scatter) into buffer 0
  U4 vst[4];
#pragma unroll
  for (int i = 0; i < 4; ++i) {
    unsigned idx = (unsigned)(tid + 256 * i);           // 1024 b128 chunks (64x128)
    async_ld_b128_s(&lds[idx * 8], kg + (size_t)kb_lo * 64 * DKV, idx * 16u);
    vst[i].q = ((const uint4*)(vg + (size_t)kb_lo * 64 * DKV))[idx];
  }
#pragma unroll
  for (int i = 0; i < 4; ++i) {
    int idx = tid + 256 * i;
    int key = idx >> 4, dv0 = (idx & 15) * 8;
#pragma unroll
    for (int e = 0; e < 8; ++e) lds[16384 + (dv0 + e) * 64 + key] = vst[i].s[e];
  }
  wait_async();
  __syncthreads();

  FragC o[8];
#pragma unroll
  for (int c = 0; c < 8; ++c)
#pragma unroll
    for (int j = 0; j < 8; ++j) o[c].f[j] = 0.0f;
  float lrun[8];
#pragma unroll
  for (int j = 0; j < 8; ++j) lrun[j] = 0.0f;

  for (int kb = kb_lo; kb <= kb_hi; ++kb) {
    const int cur = (kb - kb_lo) & 1;
    const int nxt = cur ^ 1;
    const bool more = (kb < kb_hi);
    unsigned short* lkc = lds + cur * 8192;             // current K ( / P ) buffer
    unsigned short* lvc = lds + 16384 + cur * 8192;     // current V^T buffer

    if (more) {   // prefetch next K (async->LDS) and next V (->regs)
      const unsigned short* kgn = kg + (size_t)(kb + 1) * 64 * DKV;
      const unsigned short* vgn = vg + (size_t)(kb + 1) * 64 * DKV;
      unsigned short* lkn = lds + nxt * 8192;
#pragma unroll
      for (int i = 0; i < 4; ++i) {
        unsigned idx = (unsigned)(tid + 256 * i);
        async_ld_b128_s(lkn + idx * 8, kgn, idx * 16u);
        vst[i].q = ((const uint4*)vgn)[idx];
      }
    }

    // ---- S = Q K^T  (16 WMMA/wave, 2 live B-fragments)
    FragC s[4];
#pragma unroll
    for (int c = 0; c < 4; ++c)
#pragma unroll
      for (int j = 0; j < 8; ++j) s[c].f[j] = 0.0f;
#pragma unroll
    for (int kd = 0; kd < 4; ++kd) {
#pragma unroll
      for (int cp = 0; cp < 2; ++cp) {
        Frag16 bk0, bk1;
        int n0 = (2 * cp) * 16 + lo, n1 = (2 * cp + 1) * 16 + lo;
#pragma unroll
        for (int j = 0; j < 8; ++j) {
          int k = kd * 32 + ((j & 4) << 2) + hi * 8 + ((j & 3) << 1);
          bk0.u[j] = *(const unsigned int*)&lkc[n0 * 128 + k];
          bk1.u[j] = *(const unsigned int*)&lkc[n1 * 128 + k];
        }
        s[2 * cp].v     = wmma_bf16(qf[kd].v, bk0.v, s[2 * cp].v);
        s[2 * cp + 1].v = wmma_bf16(qf[kd].v, bk1.v, s[2 * cp + 1].v);
      }
    }

    __syncthreads();   // all waves done with current K before P aliases it

    // ---- p = exp(s*scale); scores bounded by l2norm -> no running max needed.
    if (kb >= 2 * qt) {
      // diagonal blocks: per-element causal mask
#pragma unroll
      for (int c = 0; c < 4; ++c)
#pragma unroll
        for (int j = 0; j < 8; ++j) {
          int qrow = qt * 128 + wv * 16 + j + hi * 8;
          int kcol = kb * 64 + c * 16 + lo;
          s[c].f[j] = (kcol > qrow) ? 0.0f : __expf(s[c].f[j] * scl);
        }
    } else {
#pragma unroll
      for (int c = 0; c < 4; ++c)
#pragma unroll
        for (int j = 0; j < 8; ++j)
          s[c].f[j] = __expf(s[c].f[j] * scl);
    }

    // ---- P -> dead K buffer (own 16 rows only; same-wave DS ordering)
#pragma unroll
    for (int c = 0; c < 4; ++c)
#pragma unroll
      for (int j = 0; j < 8; ++j)
        lkc[(wv * 16 + j + hi * 8) * 64 + c * 16 + lo] = f2bf(s[c].f[j]);

    // ---- O += P V ; row sums l += P*ones (18 WMMA/wave, 2 live B-fragments)
    FragC z;
#pragma unroll
    for (int j = 0; j < 8; ++j) z.f[j] = 0.0f;
#pragma unroll
    for (int kd2 = 0; kd2 < 2; ++kd2) {
      Frag16 pf;
#pragma unroll
      for (int j = 0; j < 8; ++j) {
        int k = kd2 * 32 + ((j & 4) << 2) + hi * 8 + ((j & 3) << 1);
        pf.u[j] = *(const unsigned int*)&lkc[m * 64 + k];
      }
      z.v = wmma_bf16(pf.v, ones.v, z.v);               // row sums, replicated
#pragma unroll
      for (int cp = 0; cp < 4; ++cp) {
        Frag16 bv0, bv1;
        int n0 = (2 * cp) * 16 + lo, n1 = (2 * cp + 1) * 16 + lo;
#pragma unroll
        for (int j = 0; j < 8; ++j) {
          int k = kd2 * 32 + ((j & 4) << 2) + hi * 8 + ((j & 3) << 1);
          bv0.u[j] = *(const unsigned int*)&lvc[n0 * 64 + k];
          bv1.u[j] = *(const unsigned int*)&lvc[n1 * 64 + k];
        }
        o[2 * cp].v     = wmma_bf16(pf.v, bv0.v, o[2 * cp].v);
        o[2 * cp + 1].v = wmma_bf16(pf.v, bv1.v, o[2 * cp + 1].v);
      }
    }
#pragma unroll
    for (int j = 0; j < 8; ++j) lrun[j] += z.f[j];

    if (more) {   // scatter prefetched V into the other buffer
      unsigned short* lvn = lds + 16384 + nxt * 8192;
#pragma unroll
      for (int i = 0; i < 4; ++i) {
        int idx = tid + 256 * i;
        int key = idx >> 4, dv0 = (idx & 15) * 8;
#pragma unroll
        for (int e = 0; e < 8; ++e) lvn[(dv0 + e) * 64 + key] = vst[i].s[e];
      }
      wait_async();
    }
    __syncthreads();
  }

  // ---- epilogue: O / l, fp32 out [b][h][t][dv]
  float* og = out + (headoff + (size_t)qt * 128) * DKV;
#pragma unroll
  for (int j = 0; j < 8; ++j) {
    float inv = 1.0f / fmaxf(lrun[j], 1e-30f);
    int row = wv * 16 + j + hi * 8;
#pragma unroll
    for (int c = 0; c < 8; ++c)
      og[(size_t)row * DKV + c * 16 + lo] = o[c].f[j] * inv;
  }
}

// ---------------------------------------------------------------- launch
extern "C" void kernel_launch(void* const* d_in, const int* in_sizes, int n_in,
                              void* d_out, int out_size, void* d_ws, size_t ws_size,
                              hipStream_t stream) {
  (void)in_sizes; (void)n_in; (void)out_size; (void)ws_size;

  const size_t NX = (size_t)BATCH * TSEQ * DIN;
  const size_t NW = (size_t)NHEAD * DIN * DKV;
  const size_t NQ = (size_t)BATCH * NHEAD * TSEQ * DKV;

  unsigned short* xb = (unsigned short*)d_ws;
  unsigned short* wb[6];
  for (int i = 0; i < 6; ++i) wb[i] = xb + NX + (size_t)i * NW;
  unsigned short* qbuf = xb + NX + 6 * NW;
  unsigned short* kbuf = qbuf + NQ;
  unsigned short* vbuf = kbuf + NQ;

  // stage 1: fp32 -> bf16 (x plain; all 6 weights transposed in one launch)
  cvt_bf16_kernel<<<1024, 256, 0, stream>>>((const float*)d_in[0], xb, NX / 8);
  CvtArgs ca;
  for (int i = 0; i < 6; ++i) { ca.src[i] = (const float*)d_in[1 + i]; ca.dst[i] = wb[i]; }
  dim3 wgrid((DIN / 32) * (DKV / 32), NHEAD, 6);
  cvt_wT_kernel<<<wgrid, 256, 0, stream>>>(ca);

  // stage 2: q,k,v projections + L2 norm in one launch
  ProjArgs pa;
  pa.wc[0] = wb[0]; pa.wc[1] = wb[1]; pa.wc[2] = wb[2];
  pa.ws[0] = wb[3]; pa.ws[1] = wb[4]; pa.ws[2] = wb[5];
  pa.ob[0] = qbuf;  pa.ob[1] = kbuf;  pa.ob[2] = vbuf;
  dim3 pgrid(TSEQ / 128, 3 * NHEAD, BATCH);
  proj_l2_kernel<<<pgrid, 256, 0, stream>>>(xb, pa);

  // stage 3: masked flash attention (longest tiles first)
  dim3 agrid(TSEQ / 128, NHEAD, BATCH);
  attn_kernel<<<agrid, 256, 0, stream>>>(qbuf, kbuf, vbuf,
                                         (const float*)d_in[7], (float*)d_out);
}
